// MultiHeadAttentionNetwork_29918742184178
// MI455X (gfx1250) — matrix operand order repaired and checked by reading, not compiled, not run on hardware
//
#include <hip/hip_runtime.h>
#include <hip/hip_bf16.h>
#include <cstdint>
#include <cstddef>

typedef __attribute__((ext_vector_type(16))) _Float16 v16h;
typedef __attribute__((ext_vector_type(8)))  _Float16 v8h;
typedef __attribute__((ext_vector_type(8)))  float    v8f;

#define L_IN    784
#define C_DIM   128
#define KCHUNKS 25        // ceil(784/32), last chunk zero-padded in prep
#define NWAVES  8         // 8 wave32 per block, 16 rows each -> 128 rows/block

#define USE_ASYNC_LDS 1   // CDNA5 global_load_async_to_lds_* staging path

// ---------------------------------------------------------------------------
// Prep kernel: lay out W0 (f16, zero-padded, WMMA B-fragment order) and
// Wa/Wb fragments in workspace. B-frag layout per 32-deep K chunk:
//   [lane][i] : lane = 16*((k_in_chunk%16)/8) + col%16 , i = 8*(k_in_chunk/16) + k_in_chunk%8
// gW0h : [25][8 coltiles][32 lanes][16]  (204800 B)
// gWabh: [16 frags][32 lanes][16]        ( 16384 B), fr = head*8+mat*4+kc2*2+t2
// ---------------------------------------------------------------------------
__global__ __launch_bounds__(256) void mil_prep_kernel(
    const float* __restrict__ W0, const float* __restrict__ Wa,
    const float* __restrict__ Wb,
    _Float16* __restrict__ gW0h, _Float16* __restrict__ gWabh)
{
  const int total0 = KCHUNKS * 8 * 32 * 16;   // 102400
  const int total1 = 16 * 32 * 16;            // 8192
  for (int idx = blockIdx.x * blockDim.x + threadIdx.x;
       idx < total0 + total1; idx += gridDim.x * blockDim.x) {
    if (idx < total0) {
      int kc  = idx >> 12;           // / (8*512)
      int rem = idx & 4095;
      int t  = rem >> 9;
      int l2 = (rem >> 4) & 31;
      int i  = rem & 15;
      int k = kc * 32 + (l2 >> 4) * 8 + (i & 7) + (i >> 3) * 16;
      int n = t * 16 + (l2 & 15);
      gW0h[idx] = (k < L_IN) ? (_Float16)W0[k * C_DIM + n] : (_Float16)0.f;
    } else {
      int j  = idx - total0;
      int fr = j >> 9;
      int l2 = (j >> 4) & 31;
      int i  = j & 15;
      int head = fr >> 3, mat = (fr >> 2) & 1, kc2 = (fr >> 1) & 1, t2 = fr & 1;
      int k = kc2 * 32 + (l2 >> 4) * 8 + (i & 7) + (i >> 3) * 16;  // 0..63
      int d = t2 * 16 + (l2 & 15);            // 0..31
      const float* Wm = mat ? Wb : Wa;        // [2][64][32]
      gWabh[j] = (_Float16)Wm[head * 64 * 32 + k * 32 + d];
    }
  }
}

// ---------------------------------------------------------------------------
// Kernel 1: x = relu(h@W0+b0) (f16 WMMA, f32 acc, double-buffered async LDS
// staging of W0 fragments) + fused per-head gated attention logits.
// ---------------------------------------------------------------------------
__global__ __launch_bounds__(256) void mil_gemm_attn_kernel(
    const float* __restrict__ h, const _Float16* __restrict__ gW0h,
    const _Float16* __restrict__ gWabh,
    const float* __restrict__ b0,
    const float* __restrict__ ba, const float* __restrict__ bb,
    const float* __restrict__ Wc, const float* __restrict__ bc,
    float* __restrict__ gX, float* __restrict__ Alog, int N)
{
  __shared__ __attribute__((aligned(32))) _Float16 sW0[2][8][32][16]; // 16 KB
  __shared__ __attribute__((aligned(32))) _Float16 sX[NWAVES][16][128]; // 32 KB

  const int tid  = threadIdx.x;
  const int lane = tid & 31;
  const int wv   = tid >> 5;
  const int c    = lane & 15;   // col (B/C/D) or row (A) index
  const int hf   = lane >> 4;   // lane half

  const int tile = blockIdx.x * NWAVES + wv;
  const int m0 = tile * 16;
  const bool active = (m0 < N);       // wave-uniform
  const int rowA = m0 + c;

  // --- stage one 8KB W0 chunk into sW0[buf]: 256 threads x 32 B
  auto stage = [&](int kc, int buf) {
    const _Float16* src = gW0h + (size_t)kc * 4096 + tid * 16;
    _Float16* dst = &sW0[buf][0][0][0] + tid * 16;
#if USE_ASYNC_LDS
    uint64_t ga = (uint64_t)(uintptr_t)src;
    uint32_t la = (uint32_t)(uintptr_t)dst;   // low 32 bits = LDS byte address
    asm volatile("global_load_async_to_lds_b128 %0, %1, off"
                 :: "v"(la), "v"(ga) : "memory");
    asm volatile("global_load_async_to_lds_b128 %0, %1, off offset:16"
                 :: "v"(la), "v"(ga) : "memory");
#else
    v8h a0 = *(const v8h*)(src);
    v8h a1 = *(const v8h*)(src + 8);
    *(v8h*)(dst) = a0;
    *(v8h*)(dst + 8) = a1;
#endif
  };

  v8f acc[8] = {};

  stage(0, 0);
#if USE_ASYNC_LDS
  asm volatile("s_wait_asynccnt 0x0" ::: "memory");
#endif
  __syncthreads();

  for (int kc = 0; kc < KCHUNKS; ++kc) {
    const int buf = kc & 1;
    if (kc + 1 < KCHUNKS) stage(kc + 1, buf ^ 1);   // overlap copy with WMMA

    if (active) {
      // A fragment: row rowA, K = kb..kb+7 and kb+16..kb+23
      int kb = kc * 32 + (hf ? 8 : 0);
      const float* hp = h + (size_t)rowA * L_IN + kb;
      float4 q0 = *(const float4*)(hp);
      float4 q1 = *(const float4*)(hp + 4);
      float4 q2 = {0.f, 0.f, 0.f, 0.f}, q3 = {0.f, 0.f, 0.f, 0.f};
      if (kc < KCHUNKS - 1) {           // last chunk: K 784..799 are zero
        q2 = *(const float4*)(hp + 16);
        q3 = *(const float4*)(hp + 20);
      }
      v16h af;
      af[0]=(_Float16)q0.x;  af[1]=(_Float16)q0.y;  af[2]=(_Float16)q0.z;  af[3]=(_Float16)q0.w;
      af[4]=(_Float16)q1.x;  af[5]=(_Float16)q1.y;  af[6]=(_Float16)q1.z;  af[7]=(_Float16)q1.w;
      af[8]=(_Float16)q2.x;  af[9]=(_Float16)q2.y;  af[10]=(_Float16)q2.z; af[11]=(_Float16)q2.w;
      af[12]=(_Float16)q3.x; af[13]=(_Float16)q3.y; af[14]=(_Float16)q3.z; af[15]=(_Float16)q3.w;

      // preload all 8 B fragments, then issue WMMAs back-to-back
      v16h bf[8];
      #pragma unroll
      for (int t = 0; t < 8; ++t)
        bf[t] = *(const v16h*)(&sW0[buf][t][lane][0]);
      #pragma unroll
      for (int t = 0; t < 8; ++t)
        acc[t] = __builtin_amdgcn_wmma_f32_16x16x32_f16(
            false, af, false, bf[t], (short)0, acc[t], false, false);
    }
#if USE_ASYNC_LDS
    asm volatile("s_wait_asynccnt 0x0" ::: "memory");
#endif
    __syncthreads();
  }

  // ---- epilogue: bias + relu, write f32 x, stash f16 tile for attention
  if (active) {
    #pragma unroll
    for (int t = 0; t < 8; ++t) {
      float bias = b0[t * 16 + c];
      #pragma unroll
      for (int r = 0; r < 8; ++r) {
        float v = acc[t][r] + bias;
        v = v > 0.f ? v : 0.f;
        int rr = r + 8 * hf;
        gX[(size_t)(m0 + rr) * C_DIM + t * 16 + c] = v;
        sX[wv][rr][t * 16 + c] = (_Float16)v;
      }
    }
  }
  __syncthreads();

  // ---- gated attention logits per head (WMMA over S=64 -> D=32)
  if (active) {
    #pragma unroll
    for (int head = 0; head < 2; ++head) {
      v8f aacc[2] = {}, gacc[2] = {};
      #pragma unroll
      for (int kc2 = 0; kc2 < 2; ++kc2) {
        int kb = head * 64 + kc2 * 32 + (hf ? 8 : 0);
        const _Float16* xr = &sX[wv][c][0];
        v8h lo = *(const v8h*)(xr + kb);
        v8h hi = *(const v8h*)(xr + kb + 16);
        v16h af;
        #pragma unroll
        for (int i = 0; i < 8; ++i) { af[i] = lo[i]; af[8 + i] = hi[i]; }
        #pragma unroll
        for (int t2 = 0; t2 < 2; ++t2) {
          v16h bfa = *(const v16h*)(gWabh + (size_t)(head * 8 + kc2 * 2 + t2) * 512 + lane * 16);
          v16h bfg = *(const v16h*)(gWabh + (size_t)(head * 8 + 4 + kc2 * 2 + t2) * 512 + lane * 16);
          aacc[t2] = __builtin_amdgcn_wmma_f32_16x16x32_f16(
              false, af, false, bfa, (short)0, aacc[t2], false, false);
          gacc[t2] = __builtin_amdgcn_wmma_f32_16x16x32_f16(
              false, af, false, bfg, (short)0, gacc[t2], false, false);
        }
      }
      float pr[8] = {0.f,0.f,0.f,0.f,0.f,0.f,0.f,0.f};
      #pragma unroll
      for (int t2 = 0; t2 < 2; ++t2) {
        int d = t2 * 16 + c;
        float bav = ba[head * 32 + d];
        float bbv = bb[head * 32 + d];
        float wcv = Wc[head * 32 + d];
        #pragma unroll
        for (int r = 0; r < 8; ++r) {
          float av = tanhf(aacc[t2][r] + bav);
          float gv = 1.f / (1.f + __expf(-(gacc[t2][r] + bbv)));
          pr[r] += av * gv * wcv;
        }
      }
      float bcv = bc[head];
      #pragma unroll
      for (int r = 0; r < 8; ++r) {
        float v = pr[r];
        v += __shfl_xor(v, 1, 32);
        v += __shfl_xor(v, 2, 32);
        v += __shfl_xor(v, 4, 32);
        v += __shfl_xor(v, 8, 32);
        if (c == r) Alog[(size_t)(m0 + r + 8 * hf) * 2 + head] = v + bcv;
      }
    }
  }
}

// ---------------------------------------------------------------------------
// Kernel 2: partial max over rows of a [n,2] array -> out[block][2]
// ---------------------------------------------------------------------------
__global__ __launch_bounds__(256) void mil_max_partial_kernel(
    const float* __restrict__ A, float* __restrict__ out, int n)
{
  __shared__ float s0[256], s1[256];
  int tid = threadIdx.x;
  float m0 = -3.4e38f, m1 = -3.4e38f;
  for (int i = blockIdx.x * 256 + tid; i < n; i += gridDim.x * 256) {
    m0 = fmaxf(m0, A[(size_t)i * 2]);
    m1 = fmaxf(m1, A[(size_t)i * 2 + 1]);
  }
  s0[tid] = m0; s1[tid] = m1;
  for (int s = 128; s > 0; s >>= 1) {
    __syncthreads();
    if (tid < s) { s0[tid] = fmaxf(s0[tid], s0[tid + s]);
                   s1[tid] = fmaxf(s1[tid], s1[tid + s]); }
  }
  if (tid == 0) { out[(size_t)blockIdx.x * 2] = s0[0];
                  out[(size_t)blockIdx.x * 2 + 1] = s1[0]; }
}

// ---------------------------------------------------------------------------
// Kernel 3: partial exp-sums + softmax-weighted column sums of x
// ---------------------------------------------------------------------------
__global__ __launch_bounds__(256) void mil_pool_partial_kernel(
    const float* __restrict__ Alog, const float* __restrict__ gmax,
    const float* __restrict__ gX, float* __restrict__ pesum,
    float* __restrict__ pwsum, int N)
{
  __shared__ float se[128];      // exp weights for 64 instances x 2 heads
  __shared__ float sred[256];
  int tid = threadIdx.x;
  int chunk = (N + gridDim.x - 1) / gridDim.x;
  int nb = blockIdx.x * chunk;
  int ne = nb + chunk; if (ne > N) ne = N;
  float g0 = gmax[0], g1 = gmax[1];
  float eacc = 0.f, wacc = 0.f;
  int ccol = tid & 127;
  int grp = tid >> 7;
  for (int base = nb; base < ne; base += 64) {
    int cnt = ne - base; if (cnt > 64) cnt = 64;
    __syncthreads();
    if (tid < 128) {
      int i = tid >> 1, hh = tid & 1;
      if (i < cnt) {
        float e = __expf(Alog[(size_t)(base + i) * 2 + hh] - (hh ? g1 : g0));
        se[i * 2 + hh] = e;
        eacc += e;
      }
    }
    __syncthreads();
    int i0 = grp * 32;
    int i1 = i0 + 32; if (i1 > cnt) i1 = cnt;
    for (int i = i0; i < i1; ++i)
      wacc += se[i * 2 + (ccol >> 6)] * gX[(size_t)(base + i) * C_DIM + ccol];
  }
  __syncthreads();
  sred[tid] = wacc;
  __syncthreads();
  if (tid < 128) pwsum[(size_t)blockIdx.x * C_DIM + tid] = sred[tid] + sred[tid + 128];
  __syncthreads();
  sred[tid] = (tid < 128) ? eacc : 0.f;      // parity of tid == head
  for (int s = 128; s >= 2; s >>= 1) {
    __syncthreads();
    if (tid < s) sred[tid] += sred[tid + s];
  }
  __syncthreads();
  if (tid == 0) { pesum[(size_t)blockIdx.x * 2] = sred[0];
                  pesum[(size_t)blockIdx.x * 2 + 1] = sred[1]; }
}

// ---------------------------------------------------------------------------
// Kernel 4: final combine -> M[128] -> classifier -> sigmoid -> (prob, hat)
// ---------------------------------------------------------------------------
__global__ __launch_bounds__(128) void mil_final_kernel(
    const float* __restrict__ pesum, const float* __restrict__ pwsum,
    const float* __restrict__ Wcls, const float* __restrict__ bcls,
    float* __restrict__ out, int NB)
{
  __shared__ float sred[128];
  __shared__ float sE[2];
  int tid = threadIdx.x;
  float e0 = 0.f, e1 = 0.f;
  for (int b = tid; b < NB; b += 128) {
    e0 += pesum[(size_t)b * 2];
    e1 += pesum[(size_t)b * 2 + 1];
  }
  sred[tid] = e0;
  for (int s = 64; s > 0; s >>= 1) { __syncthreads(); if (tid < s) sred[tid] += sred[tid + s]; }
  __syncthreads();
  if (tid == 0) sE[0] = sred[0];
  __syncthreads();
  sred[tid] = e1;
  for (int s = 64; s > 0; s >>= 1) { __syncthreads(); if (tid < s) sred[tid] += sred[tid + s]; }
  __syncthreads();
  if (tid == 0) sE[1] = sred[0];
  __syncthreads();

  float ms = 0.f;
  for (int b = 0; b < NB; ++b) ms += pwsum[(size_t)b * C_DIM + tid];
  float M = ms / sE[tid >> 6];               // head = col/64
  sred[tid] = M * Wcls[tid];
  for (int s = 64; s > 0; s >>= 1) { __syncthreads(); if (tid < s) sred[tid] += sred[tid + s]; }
  __syncthreads();
  if (tid == 0) {
    float logit = sred[0] + bcls[0];
    float prob = 1.f / (1.f + __expf(-logit));
    out[0] = prob;
    out[1] = (prob >= 0.5f) ? 1.f : 0.f;
  }
}

// ---------------------------------------------------------------------------
extern "C" void kernel_launch(void* const* d_in, const int* in_sizes, int n_in,
                              void* d_out, int out_size, void* d_ws, size_t ws_size,
                              hipStream_t stream) {
  (void)n_in; (void)out_size; (void)ws_size;
  const float* h    = (const float*)d_in[0];
  const float* W0   = (const float*)d_in[1];
  const float* b0   = (const float*)d_in[2];
  const float* Wa   = (const float*)d_in[3];
  const float* ba   = (const float*)d_in[4];
  const float* Wb   = (const float*)d_in[5];
  const float* bb   = (const float*)d_in[6];
  const float* Wc   = (const float*)d_in[7];
  const float* bc   = (const float*)d_in[8];
  const float* Wcls = (const float*)d_in[9];
  const float* bcls = (const float*)d_in[10];
  float* out = (float*)d_out;

  const int N = in_sizes[0] / L_IN;
  const int NB_MAX  = 256;
  const int NB_POOL = 240;

  // f16 fragment arrays first (16B-aligned at ws base), f32 scratch after
  _Float16* gW0h  = (_Float16*)d_ws;                       // 25*4096 halfs
  _Float16* gWabh = gW0h + (size_t)KCHUNKS * 4096;         // 16*512 halfs
  float* fbase = (float*)(gWabh + 16 * 512);
  float* gX    = fbase;                                    // N*128
  float* Alog  = gX + (size_t)N * C_DIM;                   // N*2
  float* pmax  = Alog + (size_t)N * 2;                     // NB_MAX*2
  float* gmax  = pmax + (size_t)NB_MAX * 2;                // 2
  float* pesum = gmax + 2;                                 // NB_POOL*2
  float* pwsum = pesum + (size_t)NB_POOL * 2;              // NB_POOL*128

  mil_prep_kernel<<<128, 256, 0, stream>>>(W0, Wa, Wb, gW0h, gWabh);

  int tiles = (N + 15) / 16;
  int grid1 = (tiles + NWAVES - 1) / NWAVES;
  mil_gemm_attn_kernel<<<grid1, 256, 0, stream>>>(h, gW0h, gWabh, b0, ba, bb,
                                                  Wc, bc, gX, Alog, N);
  mil_max_partial_kernel<<<NB_MAX, 256, 0, stream>>>(Alog, pmax, N);
  mil_max_partial_kernel<<<1, 256, 0, stream>>>(pmax, gmax, NB_MAX);
  mil_pool_partial_kernel<<<NB_POOL, 256, 0, stream>>>(Alog, gmax, gX, pesum, pwsum, N);
  mil_final_kernel<<<1, 128, 0, stream>>>(pesum, pwsum, Wcls, bcls, out, NB_POOL);
}
